// SimpleMoE_3006477107617
// MI455X (gfx1250) — compile-verified, matrix operand order checked by reading, not run on hardware
//
#include <hip/hip_runtime.h>

// ---------------------------------------------------------------------------
// Fused dense-MoE for MI455X (gfx1250, wave32, WMMA).
//   gating: top2-masked softmax (1 wave/row)
//   per expert e: h = relu(x @ W1[e] + b1[e])          (bf16 WMMA, f32 accum)
//                 out += gate[:,e] * (h @ W2[e]+b2[e]) (bf16 WMMA, f32 accum)
// Data movement: GLOBAL_LOAD_ASYNC_TO_LDS_B128 double-buffered (ASYNCcnt),
// so no staging VGPRs live across barriers (round-1 version spilled those to
// scratch). Weights pre-transposed to [N,K] so A and B fragments are both
// contiguous 16B runs matching the CDNA5 16-bit fragment layout.
// ---------------------------------------------------------------------------

typedef __bf16 bf16_t;
typedef __attribute__((ext_vector_type(16))) __bf16 v16bf;
typedef __attribute__((ext_vector_type(8)))  float  v8f;

static constexpr int Bsz = 8192;
static constexpr int Din = 1024;
static constexpr int Hid = 2048;
static constexpr int Onn = 1024;
static constexpr int NE  = 8;

static constexpr int BM = 128, BN = 128, BK = 32;
static constexpr int BKP = 40;   // padded LDS row (80B) -> conflict-free frag reads

__device__ __forceinline__ bf16_t to_bf16(float f) {
  unsigned u = __float_as_uint(f);
  unsigned r = u + 0x7FFFu + ((u >> 16) & 1u);   // round-to-nearest-even
  unsigned short h = (unsigned short)(r >> 16);
  return __builtin_bit_cast(bf16_t, h);
}

// ---- CDNA5 async global->LDS copy (16B per lane), with safe fallback ------
__device__ __forceinline__ void async_copy16(const void* g, void* l) {
#if __has_builtin(__builtin_amdgcn_global_load_async_to_lds_b128)
  // Signature (from hipcc diagnostic): (v4i AS1*, v4i AS3*, Imm offset, Imm cpol)
  typedef int v4i_t __attribute__((ext_vector_type(4)));
  typedef __attribute__((address_space(1))) v4i_t* gp_t;
  typedef __attribute__((address_space(3))) v4i_t* lp_t;
  __builtin_amdgcn_global_load_async_to_lds_b128((gp_t)g, (lp_t)l, 0, 0);
#else
  *reinterpret_cast<uint4*>(l) = *reinterpret_cast<const uint4*>(g);
#endif
}

__device__ __forceinline__ void wait_async0() {
#if __has_builtin(__builtin_amdgcn_s_wait_asynccnt)
  __builtin_amdgcn_s_wait_asynccnt(0);
#else
  asm volatile("s_wait_asynccnt 0" ::: "memory");
#endif
}

// ---------------- gating: scores, top-2 masked softmax, renorm -------------
__global__ __launch_bounds__(256) void gating_kernel(
    const float* __restrict__ x, const float* __restrict__ Wg,
    const float* __restrict__ bg, float* __restrict__ gate) {
  const int row  = (blockIdx.x * blockDim.x + threadIdx.x) >> 5;
  const int lane = threadIdx.x & 31;
  const float* xr = x + (size_t)row * Din;
  float acc[NE];
#pragma unroll
  for (int e = 0; e < NE; ++e) acc[e] = 0.f;
  for (int d = lane; d < Din; d += 32) {
    const float xv = xr[d];
    const float* wr = Wg + (size_t)d * NE;
#pragma unroll
    for (int e = 0; e < NE; ++e) acc[e] = fmaf(xv, wr[e], acc[e]);
  }
#pragma unroll
  for (int off = 16; off > 0; off >>= 1) {
#pragma unroll
    for (int e = 0; e < NE; ++e) acc[e] += __shfl_down(acc[e], off, 32);
  }
  if (lane == 0) {
    float s[NE];
#pragma unroll
    for (int e = 0; e < NE; ++e) s[e] = acc[e] + bg[e];
    int i1 = 0;
#pragma unroll
    for (int e = 1; e < NE; ++e) if (s[e] > s[i1]) i1 = e;
    int i2 = (i1 == 0) ? 1 : 0;
#pragma unroll
    for (int e = 0; e < NE; ++e) if (e != i1 && s[e] > s[i2]) i2 = e;
    const float mx = s[i1];
    float p[NE], Z = 0.f;
#pragma unroll
    for (int e = 0; e < NE; ++e) { p[e] = __expf(s[e] - mx); Z += p[e]; }
    const float inv = 1.f / Z;
    const float den = (p[i1] + p[i2]) * inv + 1e-8f;
#pragma unroll
    for (int e = 0; e < NE; ++e)
      gate[(size_t)row * NE + e] = (e == i1 || e == i2) ? (p[e] * inv) / den : 0.f;
  }
}

// ---------------- fp32 -> bf16 elementwise ---------------------------------
__global__ void cvt_bf16_kernel(const float* __restrict__ in,
                                bf16_t* __restrict__ out, int n) {
  int i = blockIdx.x * blockDim.x + threadIdx.x;
  if (i < n) out[i] = to_bf16(in[i]);
}

// ---------------- fp32 [E,R,C] -> bf16 [E,C,R] (tiled transpose) -----------
__global__ __launch_bounds__(256) void transpose_cvt_kernel(
    const float* __restrict__ in, bf16_t* __restrict__ out, int R, int C) {
  __shared__ bf16_t tile[32][33];
  const int e = blockIdx.z;
  const float* src = in  + (size_t)e * R * C;
  bf16_t*      dst = out + (size_t)e * R * C;
  const int c0 = blockIdx.x * 32, r0 = blockIdx.y * 32;
  const int tx = threadIdx.x, ty = threadIdx.y;
#pragma unroll
  for (int i = 0; i < 4; ++i)
    tile[ty + i * 8][tx] = to_bf16(src[(size_t)(r0 + ty + i * 8) * C + (c0 + tx)]);
  __syncthreads();
#pragma unroll
  for (int i = 0; i < 4; ++i)
    dst[(size_t)(c0 + ty + i * 8) * R + (r0 + tx)] = tile[tx][ty + i * 8];
}

__global__ void zero_kernel(float* __restrict__ p, int n) {
  int i = blockIdx.x * blockDim.x + threadIdx.x;
  if (i < n) p[i] = 0.f;
}

// ---------------- bf16 WMMA GEMM, 128x128x32 tiles, 8 waves ----------------
// MODE 0: HOut = relu(A@Bt^T + bias)          as bf16   (GEMM1)
// MODE 1: FOut += gate[row,e] * (A@Bt^T+bias) as f32    (GEMM2 + combine)
template <int MODE>
__global__ __launch_bounds__(256) void moe_gemm_kernel(
    const bf16_t* __restrict__ A,    // [M,K]
    const bf16_t* __restrict__ Bt,   // [N,K] (weights transposed)
    const float*  __restrict__ bias, // [N]
    const float*  __restrict__ gate, // [B,NE] (MODE 1)
    int expert,
    bf16_t* __restrict__ HOut,       // MODE 0
    float*  __restrict__ FOut,       // MODE 1
    int M, int N, int K) {
  __shared__ bf16_t As[2][BM][BKP];
  __shared__ bf16_t Bs[2][BN][BKP];

  const int tid = threadIdx.x;
  const int m0 = blockIdx.y * BM;
  const int n0 = blockIdx.x * BN;

  const int wid  = tid >> 5;
  const int lane = tid & 31;
  const int wm = wid & 3;         // 4 waves along M -> 32 rows each
  const int wn = wid >> 2;        // 2 waves along N -> 64 cols each
  const int rBase = wm * 32;
  const int cBase = wn * 64;
  const int r  = lane & 15;
  const int hi = lane >> 4;

  // Each thread copies 2 x 16B chunks of A and of B per tile.
  int rowL[2], blkL[2];
#pragma unroll
  for (int i = 0; i < 2; ++i) { int c = tid + i * 256; rowL[i] = c >> 2; blkL[i] = c & 3; }

  v8f acc[2][4];
#pragma unroll
  for (int m = 0; m < 2; ++m)
#pragma unroll
    for (int n = 0; n < 4; ++n)
#pragma unroll
      for (int q = 0; q < 8; ++q) acc[m][n][q] = 0.f;

  // async copies: global tile kt -> LDS buffer buf (no staging VGPRs)
  auto issue_copies = [&](int buf, int kt) {
    const int k0 = kt * BK;
#pragma unroll
    for (int i = 0; i < 2; ++i) {
      const bf16_t* ga = A  + (size_t)(m0 + rowL[i]) * K + (k0 + blkL[i] * 8);
      const bf16_t* gb = Bt + (size_t)(n0 + rowL[i]) * K + (k0 + blkL[i] * 8);
      async_copy16(ga, &As[buf][rowL[i]][blkL[i] * 8]);
      async_copy16(gb, &Bs[buf][rowL[i]][blkL[i] * 8]);
    }
  };

  const int KT = K / BK;
  issue_copies(0, 0);
  wait_async0();
  __syncthreads();

  for (int kt = 0; kt < KT; ++kt) {
    const int buf = kt & 1;
    // Safe to write buf^1: barrier at end of previous iteration guarantees
    // every wave finished reading it. Copies overlap this tile's WMMAs.
    if (kt + 1 < KT) issue_copies(buf ^ 1, kt + 1);

    union UU { v16bf v; uint4 q[2]; };
    v16bf af[2], bfv[4];
#pragma unroll
    for (int m = 0; m < 2; ++m) {
      UU u;
      const uint4* p = reinterpret_cast<const uint4*>(&As[buf][rBase + m * 16 + r][hi * 8]);
      u.q[0] = p[0]; u.q[1] = p[2];   // K runs: hi*8..+7 and 16+hi*8..+7
      af[m] = u.v;
    }
#pragma unroll
    for (int n = 0; n < 4; ++n) {
      UU u;
      const uint4* p = reinterpret_cast<const uint4*>(&Bs[buf][cBase + n * 16 + r][hi * 8]);
      u.q[0] = p[0]; u.q[1] = p[2];
      bfv[n] = u.v;
    }
#pragma unroll
    for (int m = 0; m < 2; ++m)
#pragma unroll
      for (int n = 0; n < 4; ++n)
        acc[m][n] = __builtin_amdgcn_wmma_f32_16x16x32_bf16(
            false, af[m], false, bfv[n], (short)0, acc[m][n], false, false);

    // one wait + one barrier per K-step:
    //  - my async copies into buf^1 have landed
    //  - all waves are done reading buf (so next iter may overwrite it)
    wait_async0();
    __syncthreads();
  }

  // C/D layout: VGPR j -> row j + hi*8 ; col = lane%16
  if constexpr (MODE == 0) {
#pragma unroll
    for (int n = 0; n < 4; ++n) {
      const int gc = n0 + cBase + n * 16 + r;
      const float bv = bias[gc];
#pragma unroll
      for (int m = 0; m < 2; ++m) {
        const int grb = m0 + rBase + m * 16 + hi * 8;
#pragma unroll
        for (int j = 0; j < 8; ++j) {
          float v = acc[m][n][j] + bv;
          v = fmaxf(v, 0.f);
          HOut[(size_t)(grb + j) * N + gc] = to_bf16(v);
        }
      }
    }
  } else {
    float wrow[2][8];
#pragma unroll
    for (int m = 0; m < 2; ++m) {
      const int grb = m0 + rBase + m * 16 + hi * 8;
#pragma unroll
      for (int j = 0; j < 8; ++j)
        wrow[m][j] = gate[(size_t)(grb + j) * NE + expert];
    }
#pragma unroll
    for (int n = 0; n < 4; ++n) {
      const int gc = n0 + cBase + n * 16 + r;
      const float bv = bias[gc];
#pragma unroll
      for (int m = 0; m < 2; ++m) {
        const int grb = m0 + rBase + m * 16 + hi * 8;
#pragma unroll
        for (int j = 0; j < 8; ++j) {
          const size_t idx = (size_t)(grb + j) * N + gc;
          FOut[idx] = FOut[idx] + wrow[m][j] * (acc[m][n][j] + bv);
        }
      }
    }
  }
}

// ---------------------------------------------------------------------------
extern "C" void kernel_launch(void* const* d_in, const int* in_sizes, int n_in,
                              void* d_out, int out_size, void* d_ws, size_t ws_size,
                              hipStream_t stream) {
  (void)in_sizes; (void)n_in; (void)out_size; (void)ws_size;
  const float* x  = (const float*)d_in[0];
  const float* Wg = (const float*)d_in[1];
  const float* bg = (const float*)d_in[2];
  const float* W1 = (const float*)d_in[3];
  const float* b1 = (const float*)d_in[4];
  const float* W2 = (const float*)d_in[5];
  const float* b2 = (const float*)d_in[6];
  float* out = (float*)d_out;

  char* w = (char*)d_ws;
  auto alloc = [&](size_t bytes) {
    char* p = w; w += (bytes + 255) & ~(size_t)255; return p;
  };
  float*  gate = (float*) alloc((size_t)Bsz * NE  * 4);      // 0.25 MB
  bf16_t* x_bf = (bf16_t*)alloc((size_t)Bsz * Din * 2);      // 16 MB
  bf16_t* W1t  = (bf16_t*)alloc((size_t)NE * Din * Hid * 2); // 32 MB [E,H,D]
  bf16_t* W2t  = (bf16_t*)alloc((size_t)NE * Hid * Onn * 2); // 32 MB [E,O,H]
  bf16_t* h_bf = (bf16_t*)alloc((size_t)Bsz * Hid * 2);      // 32 MB

  gating_kernel<<<Bsz / 8, 256, 0, stream>>>(x, Wg, bg, gate);
  {
    const int n = Bsz * Din;
    cvt_bf16_kernel<<<(n + 255) / 256, 256, 0, stream>>>(x, x_bf, n);
  }
  transpose_cvt_kernel<<<dim3(Hid / 32, Din / 32, NE), dim3(32, 8), 0, stream>>>(W1, W1t, Din, Hid);
  transpose_cvt_kernel<<<dim3(Onn / 32, Hid / 32, NE), dim3(32, 8), 0, stream>>>(W2, W2t, Hid, Onn);
  {
    const int n = Bsz * Onn;
    zero_kernel<<<(n + 255) / 256, 256, 0, stream>>>(out, n);
  }

  for (int e = 0; e < NE; ++e) {
    moe_gemm_kernel<0><<<dim3(Hid / BN, Bsz / BM), 256, 0, stream>>>(
        x_bf, W1t + (size_t)e * Hid * Din, b1 + (size_t)e * Hid,
        nullptr, e, h_bf, nullptr, Bsz, Hid, Din);
    moe_gemm_kernel<1><<<dim3(Onn / BN, Bsz / BM), 256, 0, stream>>>(
        h_bf, W2t + (size_t)e * Onn * Hid, b2 + (size_t)e * Onn,
        gate, e, nullptr, out, Bsz, Onn, Hid);
  }
}
